// MSDeformAttn_STATrans_51582557225482
// MI455X (gfx1250) — matrix operand order repaired and checked by reading, not compiled, not run on hardware
//
#include <hip/hip_runtime.h>
#include <math.h>

// Problem constants (from reference)
#define NB   4
#define LQT  5440      // total queries (= sum of level HW)
#define DIM  256
#define NHD  8         // heads
#define NLV  4         // levels
#define NPT  4         // points
#define HDM  32        // head dim

typedef __attribute__((ext_vector_type(16))) __bf16 v16bf;
typedef __attribute__((ext_vector_type(8)))  __bf16 bf16x8;
typedef __attribute__((ext_vector_type(8)))  float  v8f;
typedef __attribute__((ext_vector_type(4)))  float  f32x4;
typedef __attribute__((ext_vector_type(4)))  unsigned int u32x4;
typedef __attribute__((ext_vector_type(8)))  int i32x8;
typedef __attribute__((ext_vector_type(4)))  int i32x4;

__device__ __forceinline__ __bf16 f2bf(float f) {
  unsigned u = __builtin_bit_cast(unsigned, f);
  u += 0x7FFFu + ((u >> 16) & 1u);          // round to nearest even
  unsigned short s = (unsigned short)(u >> 16);
  __bf16 r;
  __builtin_memcpy(&r, &s, sizeof(r));
  return r;
}

__device__ __forceinline__ v8f wmma_bf16(v16bf a, v16bf b, v8f c) {
  return __builtin_amdgcn_wmma_f32_16x16x32_bf16(false, a, false, b,
                                                 (short)0, c, false, false);
}

// Two 16B loads -> one 16-element bf16 fragment (global or LDS pointer)
__device__ __forceinline__ v16bf frag2(const __bf16* p0, const __bf16* p1) {
  bf16x8 lo = *(const bf16x8*)p0;
  bf16x8 hi = *(const bf16x8*)p1;
  return __builtin_shufflevector(lo, hi, 0, 1, 2, 3, 4, 5, 6, 7,
                                 8, 9, 10, 11, 12, 13, 14, 15);
}

// ---------------------------------------------------------------------------
// f32 -> bf16 conversion pass (n must be a multiple of 8)
// ---------------------------------------------------------------------------
__global__ __launch_bounds__(256)
void cvt_f32_bf16(const float* __restrict__ in, __bf16* __restrict__ out,
                  unsigned n) {
  unsigned i = ((unsigned)blockIdx.x * blockDim.x + threadIdx.x) * 8u;
  if (i >= n) return;
  f32x4 v0 = *(const f32x4*)(in + i);
  f32x4 v1 = *(const f32x4*)(in + i + 4);
  bf16x8 o;
#pragma unroll
  for (int j = 0; j < 4; ++j) { o[j] = f2bf(v0[j]); o[j + 4] = f2bf(v1[j]); }
  *(bf16x8*)(out + i) = o;
}

// ---------------------------------------------------------------------------
// C[M,Ncols] = A[M,K]bf16 @ W[Ncols,ldw]bf16^T (cols at +wofs) (+bias) (+Cin)
//
// Block = 8 waves covering (8/TILESN) M-tiles x TILESN N-tiles; the block's
// A rows (32*8/TILESN x K bf16) are DMA'd into LDS once by the Tensor Data
// Mover (tensor_load_to_lds) with a 16B-per-row pad so ds_load_b128 reads are
// bank-conflict-free. Each wave32 owns a 32x32 tile: 4 accumulators, 4
// v_wmma_f32_16x16x32_bf16 per fully-unrolled K step; B streams from global
// (weights are L2-resident).
// ---------------------------------------------------------------------------
template<int K, int TILESN, bool HAS_BIAS, bool HAS_CIN>
__global__ __launch_bounds__(256)
void wmma_gemm_bf16(const __bf16* __restrict__ A,
                    const __bf16* __restrict__ W, int ldw, int wofs,
                    const float* __restrict__ bias,
                    const float* __restrict__ Cin,
                    float* __restrict__ C,
                    int M, int Ncols) {
  constexpr int MT_PER_BLK = 8 / TILESN;
  constexpr int ROWS_PER_BLK = MT_PER_BLK * 32;
  constexpr int LDS_STRIDE = K + 8;          // +16B pad per 512B row
  __shared__ __bf16 smemA[ROWS_PER_BLK * LDS_STRIDE];

  const int waveInBlk = threadIdx.x >> 5;
  const int mtLocal = waveInBlk / TILESN;
  const int nt = waveInBlk - mtLocal * TILESN;
  const int mt = blockIdx.x * MT_PER_BLK + mtLocal;
  const int lane = threadIdx.x & 31;
  const int half = lane >> 4;          // 0: lanes 0-15, 1: lanes 16-31
  const int l16  = lane & 15;

  // ---- TDM: DMA this block's A tile (ROWS_PER_BLK x K bf16) into LDS ----
  if (waveInBlk == 0) {
    const unsigned long long ga =
        (unsigned long long)(const char*)(A + (size_t)blockIdx.x * ROWS_PER_BLK * K);
    const unsigned ldsb = (unsigned)(unsigned long long)(&smemA[0]);
    u32x4 g0;
    g0[0] = 1u;                                   // count=1, user descriptor
    g0[1] = ldsb;                                 // lds_addr (bytes)
    g0[2] = (unsigned)(ga & 0xFFFFFFFFu);         // global_addr[31:0]
    g0[3] = (unsigned)((ga >> 32) & 0x1FFFFFFu) | (2u << 30);  // addr[56:32], type=2
    i32x8 g1;
    // data_size=2B (code 1); pad_enable; pad_interval=6 (128 DW = 512B);
    // pad_amount=3 (4 DW = 16B)
    g1[0] = (1 << 16) | (1 << 20) | (6 << 22) | (3 << 25);
    g1[1] = (K & 0xFFFF) << 16;                            // tensor_dim0 lo16
    g1[2] = ((K >> 16) & 0xFFFF) | ((M & 0xFFFF) << 16);   // dim0 hi / dim1 lo
    g1[3] = ((M >> 16) & 0xFFFF) | ((K & 0xFFFF) << 16);   // dim1 hi / tile_dim0
    g1[4] = ROWS_PER_BLK;                                  // tile_dim1 (tile_dim2=0)
    g1[5] = K;                                             // dim0_stride lo32
    g1[6] = 0;
    g1[7] = 0;
    i32x4 gz4 = {0, 0, 0, 0};
    i32x8 gz8 = {0, 0, 0, 0, 0, 0, 0, 0};
    __builtin_amdgcn_tensor_load_to_lds(g0, g1, gz4, gz4, gz8, 0);
    __builtin_amdgcn_s_wait_tensorcnt(0);
  }
  __syncthreads();

  // CDNA5 wave32 WMMA fragment layout (ISA 7.12.2):
  //  A 16x32: lane holds row (lane&15); K chunks {0..7,16..23} lo-half,
  //           {8..15,24..31} hi-half.
  //  B 32x16: lane holds col (lane&15); same K chunks as A.
  const int aLo = half ? 8 : 0;
  const int aHi = half ? 24 : 16;
  const int bK  = half ? 16 : 0;

  const __bf16* aLds0 = &smemA[((mtLocal << 5) + l16) * LDS_STRIDE];
  const __bf16* aLds1 = aLds0 + 16 * LDS_STRIDE;
  const __bf16* wrow0 = W + (size_t)((nt << 5) + l16) * ldw + wofs;
  const __bf16* wrow1 = wrow0 + (size_t)16 * ldw;

  v8f acc00 = {}, acc01 = {}, acc10 = {}, acc11 = {};
#pragma unroll
  for (int k0 = 0; k0 < K; k0 += 32) {
    v16bf a0 = frag2(aLds0 + k0 + aLo, aLds0 + k0 + aHi);
    v16bf a1 = frag2(aLds1 + k0 + aLo, aLds1 + k0 + aHi);
    v16bf b0 = frag2(wrow0 + k0 + aLo, wrow0 + k0 + aHi);
    v16bf b1 = frag2(wrow1 + k0 + aLo, wrow1 + k0 + aHi);
    acc00 = wmma_bf16(a0, b0, acc00);
    acc01 = wmma_bf16(a0, b1, acc01);
    acc10 = wmma_bf16(a1, b0, acc10);
    acc11 = wmma_bf16(a1, b1, acc11);
  }

  const int col0 = (nt << 5) + l16;
  const int col1 = col0 + 16;
  float badd0 = 0.f, badd1 = 0.f;
  if constexpr (HAS_BIAS) { badd0 = bias[col0]; badd1 = bias[col1]; }

#pragma unroll
  for (int mi = 0; mi < 2; ++mi) {
    const v8f& aL = mi ? acc10 : acc00;
    const v8f& aR = mi ? acc11 : acc01;
#pragma unroll
    for (int r = 0; r < 8; ++r) {
      // D layout: VGPR r -> M = r (lanes 0-15) / M = 8+r (lanes 16-31)
      const int row = (mt << 5) + (mi << 4) + r + (half << 3);
      const size_t i0 = (size_t)row * Ncols + col0;
      const size_t i1 = (size_t)row * Ncols + col1;
      float v0 = aL[r] + badd0;
      float v1 = aR[r] + badd1;
      if constexpr (HAS_CIN) { v0 += Cin[i0]; v1 += Cin[i1]; }
      C[i0] = v0;
      C[i1] = v1;
    }
  }
}

// ---------------------------------------------------------------------------
// Deformable sampling: one wave per (n,q,h); lane = head-dim channel.
// Gathers are 128B fully-coalesced; `value` (22MB) is L2-resident (192MB L2).
// ---------------------------------------------------------------------------
__global__ __launch_bounds__(256)
void msda_sample(const float* __restrict__ value,   // (NB, 5440, NHD, HDM)
                 const float* __restrict__ off,     // (NB*LQT, 256)
                 const float* __restrict__ attn,    // (NB*LQT, 128)
                 const float* __restrict__ ref,     // (NB, LQT, NLV, 2)
                 float* __restrict__ acc_out)       // (NB*LQT, 256)
{
  const int wave = (blockIdx.x * blockDim.x + threadIdx.x) >> 5;
  if (wave >= NB * LQT * NHD) return;
  const int h    = wave & (NHD - 1);
  const int nq   = wave >> 3;          // n*LQT + q
  const int n    = nq / LQT;
  const int lane = threadIdx.x & 31;   // channel d

  const int lvlH[NLV]   = {64, 32, 16, 8};
  const int lvlW[NLV]   = {64, 32, 16, 8};
  const int lvlOfs[NLV] = {0, 4096, 5120, 5376};

  // softmax over 16 logits (redundant per lane; cheap vs gather latency)
  const float* ap = attn + (size_t)nq * 128 + h * 16;
  float lg[16];
  float mx = -1e30f;
#pragma unroll
  for (int i = 0; i < 16; ++i) { lg[i] = ap[i]; mx = fmaxf(mx, lg[i]); }
  float ssum = 0.f;
#pragma unroll
  for (int i = 0; i < 16; ++i) { lg[i] = __expf(lg[i] - mx); ssum += lg[i]; }
  const float inv = 1.0f / ssum;

  const float* op = off + (size_t)nq * 256 + h * 32;
  const float* rp = ref + (size_t)nq * (NLV * 2);

  float acc = 0.f;
#pragma unroll
  for (int l = 0; l < NLV; ++l) {
    const int H = lvlH[l], W = lvlW[l];
    const float rx = rp[l * 2 + 0];
    const float ry = rp[l * 2 + 1];
    const float* vbase =
        value + ((size_t)(n * LQT + lvlOfs[l]) * NHD + h) * HDM + lane;
#pragma unroll
    for (int p = 0; p < NPT; ++p) {
      const float ox = op[l * 8 + p * 2 + 0];
      const float oy = op[l * 8 + p * 2 + 1];
      const float aw = lg[l * 4 + p] * inv;
      const float x = (rx + ox / (float)W) * (float)W - 0.5f;
      const float y = (ry + oy / (float)H) * (float)H - 0.5f;
      const float xf = floorf(x), yf = floorf(y);
      const float lx = x - xf, ly = y - yf;
      const int x0 = (int)xf, y0 = (int)yf;
      float s = 0.f;
#pragma unroll
      for (int c = 0; c < 4; ++c) {
        const int xi = x0 + (c & 1);
        const int yi = y0 + (c >> 1);
        const float wx = (c & 1) ? lx : 1.f - lx;
        const float wy = (c >> 1) ? ly : 1.f - ly;
        if (xi >= 0 && xi < W && yi >= 0 && yi < H) {
          s += wx * wy * vbase[(size_t)(yi * W + xi) * (NHD * HDM)];
        }
      }
      acc += aw * s;
    }
  }
  acc_out[(size_t)nq * 256 + h * HDM + lane] = acc;
}

// ---------------------------------------------------------------------------
extern "C" void kernel_launch(void* const* d_in, const int* in_sizes, int n_in,
                              void* d_out, int out_size, void* d_ws, size_t ws_size,
                              hipStream_t stream) {
  (void)in_sizes; (void)n_in; (void)out_size; (void)ws_size;
  const float* query0 = (const float*)d_in[0];
  const float* query1 = (const float*)d_in[1];
  const float* feat0  = (const float*)d_in[2];
  const float* feat1  = (const float*)d_in[3];
  const float* refpts = (const float*)d_in[4];
  const float* W_off  = (const float*)d_in[5];
  const float* b_off  = (const float*)d_in[6];
  const float* W_attn = (const float*)d_in[7];
  const float* b_attn = (const float*)d_in[8];
  const float* W_val  = (const float*)d_in[9];
  const float* b_val  = (const float*)d_in[10];
  const float* W_out  = (const float*)d_in[11];
  const float* b_out  = (const float*)d_in[12];
  const float* W_agg  = (const float*)d_in[13];
  const float* b_agg  = (const float*)d_in[14];
  float* out = (float*)d_out;

  const size_t M = (size_t)NB * LQT;   // 21760 rows, divisible by 32
  float* ws       = (float*)d_ws;
  float* buf_val  = ws;                  // M*256  (reused as proj buffer later)
  float* buf_off  = buf_val  + M * 256;  // M*256
  float* buf_attn = buf_off  + M * 256;  // M*128
  float* buf_acc  = buf_attn + M * 128;  // M*256
  __bf16* abf  = (__bf16*)(buf_acc + M * 256);  // M*256 bf16 activation staging
  __bf16* wvb  = abf  + M * 256;   // 256*256
  __bf16* wob  = wvb  + 256 * 256; // 256*256
  __bf16* wab  = wob  + 256 * 256; // 128*256
  __bf16* wtb  = wab  + 128 * 256; // 256*256
  __bf16* wgb  = wtb  + 256 * 256; // 256*512      total ~90 MB

  auto cvt = [&](const float* src, __bf16* dst, size_t n) {
    hipLaunchKernelGGL(cvt_f32_bf16, dim3((unsigned)((n / 8 + 255) / 256)),
                       dim3(256), 0, stream, src, dst, (unsigned)n);
  };
  // exact grids: 8 waves per block cover (8/tilesN) M-tiles x tilesN N-tiles
  const dim3 grid256((unsigned)(M / 32));        // tilesN=8 -> 680 blocks
  const dim3 grid128((unsigned)(M / 64));        // tilesN=4 -> 340 blocks

  // Weights -> bf16 (cheap; done every call to stay stateless/deterministic)
  cvt(W_val,  wvb, 256 * 256);
  cvt(W_off,  wob, 256 * 256);
  cvt(W_attn, wab, 128 * 256);
  cvt(W_out,  wtb, 256 * 256);
  cvt(W_agg,  wgb, 256 * 512);

  const int sampWaves  = NB * LQT * NHD;            // 174080
  const dim3 sampGrid((sampWaves * 32 + 255) / 256);

  for (int s = 0; s < 2; ++s) {
    const float* q = s ? query1 : query0;
    const float* f = s ? feat1  : feat0;

    // value = feat @ W_val^T + b_val
    cvt(f, abf, M * 256);
    hipLaunchKernelGGL((wmma_gemm_bf16<256, 8, true, false>), grid256,
                       dim3(256), 0, stream,
                       abf, wvb, 256, 0, b_val, (const float*)nullptr,
                       buf_val, (int)M, 256);

    // off = q @ W_off^T + b_off ; attn = q @ W_attn^T + b_attn
    cvt(q, abf, M * 256);
    hipLaunchKernelGGL((wmma_gemm_bf16<256, 8, true, false>), grid256,
                       dim3(256), 0, stream,
                       abf, wob, 256, 0, b_off, (const float*)nullptr,
                       buf_off, (int)M, 256);
    hipLaunchKernelGGL((wmma_gemm_bf16<256, 4, true, false>), grid128,
                       dim3(256), 0, stream,
                       abf, wab, 256, 0, b_attn, (const float*)nullptr,
                       buf_attn, (int)M, 128);

    // softmax + bilinear deformable sampling -> acc
    hipLaunchKernelGGL(msda_sample, sampGrid, dim3(256), 0, stream,
                       buf_val, buf_off, buf_attn, refpts, buf_acc);

    // proj = acc @ W_out^T + b_out   (reuse buf_val, value no longer needed)
    cvt(buf_acc, abf, M * 256);
    hipLaunchKernelGGL((wmma_gemm_bf16<256, 8, true, false>), grid256,
                       dim3(256), 0, stream,
                       abf, wtb, 256, 0, b_out, (const float*)nullptr,
                       buf_val, (int)M, 256);

    // out (s=0): proj @ W_agg[:, :256]^T + b_agg
    // out (s=1): out += proj @ W_agg[:, 256:]^T
    cvt(buf_val, abf, M * 256);
    if (s == 0) {
      hipLaunchKernelGGL((wmma_gemm_bf16<256, 8, true, false>), grid256,
                         dim3(256), 0, stream,
                         abf, wgb, 512, 0, b_agg, (const float*)nullptr,
                         out, (int)M, 256);
    } else {
      hipLaunchKernelGGL((wmma_gemm_bf16<256, 8, false, true>), grid256,
                         dim3(256), 0, stream,
                         abf, wgb, 512, 256, (const float*)nullptr,
                         (const float*)out, out, (int)M, 256);
    }
  }
}
